// WindowAttention_conv2qkv_57758720196960
// MI455X (gfx1250) — compile-verified, matrix operand order checked by reading, not run on hardware
//
#include <hip/hip_runtime.h>
#include <hip/hip_bf16.h>

typedef __attribute__((ext_vector_type(16))) __bf16          v16bf;
typedef __attribute__((ext_vector_type(8)))  float           v8f;
typedef __attribute__((ext_vector_type(8)))  unsigned short  v8us;
typedef __attribute__((ext_vector_type(16))) unsigned short  v16us;

#define XS_BYTES   131072              // 16*16*256 bf16 halo'd x tile
#define SMEM_BYTES (131072 + 150528)   // + qkv f32 [768*49]

static __device__ __forceinline__ unsigned short f2bf(float f) {
    unsigned int u = __builtin_bit_cast(unsigned int, f);
    unsigned int r = (u + 0x7FFFu + ((u >> 16) & 1u)) >> 16;   // RNE
    return (unsigned short)r;
}

static __device__ __forceinline__ v16bf mkfrag(v8us lo, v8us hi) {
    v16us t = __builtin_shufflevector(lo, hi, 0,1,2,3,4,5,6,7,8,9,10,11,12,13,14,15);
    return __builtin_bit_cast(v16bf, t);
}

// ---------------------------------------------------------------------------
// Kernel 0: repack weights fp32 OIHW -> bf16 [oc][ (kh*3+kw)*256 + ic ]
// ---------------------------------------------------------------------------
__global__ void prep_w_kernel(const float* __restrict__ w, unsigned short* __restrict__ wB) {
    int idx = blockIdx.x * 256 + threadIdx.x;
    if (idx >= 768 * 2304) return;
    int oc = idx / 2304, rem = idx - oc * 2304;
    int g = rem >> 8, ic = rem & 255;
    wB[idx] = f2bf(w[oc * 2304 + ic * 9 + g]);
}

// ---------------------------------------------------------------------------
// Kernel 1: fused qkv-conv (implicit GEMM, bf16 WMMA, dual-M blocking)
//           + window attention. One block (8 wave32) per batch image.
//           out7 = [B][256][49] f32
// ---------------------------------------------------------------------------
__global__ void __launch_bounds__(256)
fused_qkv_attn_kernel(const float* __restrict__ x,
                      const unsigned short* __restrict__ wB,
                      const float* __restrict__ rpb,
                      float* __restrict__ out7)
{
    extern __shared__ char smem[];
    unsigned short* xs  = (unsigned short*)smem;            // [16][16][256] bf16 (halo'd HWC)
    float*          qkv = (float*)(smem + XS_BYTES);        // [768*49] f32 (flat, torch order)

    const int b    = blockIdx.x;
    const int tid  = threadIdx.x;
    const int wave = tid >> 5;
    const int lane = tid & 31;
    const int l15  = lane & 15;
    const int loff = (lane & 16) ? 8 : 0;   // K-half offset for 16-bit A/B fragments
    const int hi8  = (lane & 16) ? 8 : 0;   // M-row offset for f32 C/D fragments

    // ---- stage x[b] into LDS as bf16 HWC with zero halo --------------------
    #pragma unroll 4
    for (int cell = 0; cell < 256; ++cell) {               // zero halo rows/cols
        int ih = cell >> 4, iw = cell & 15;
        if (ih == 0 || ih == 15 || iw == 0 || iw == 15)
            xs[cell * 256 + tid] = 0;
    }
    const float* xb = x + (size_t)b * 50176;               // 256*14*14
    #pragma unroll 4
    for (int s = 0; s < 196; ++s) {
        int ih = s / 14 + 1, iw = s % 14 + 1;              // +1 halo shift (pad=1)
        xs[(ih * 16 + iw) * 256 + tid] = f2bf(xb[tid * 196 + s]);
    }
    __syncthreads();

    // per-lane B-fragment base addresses (hoisted: kc part is wave-uniform)
    int bbase[4];
    #pragma unroll
    for (int nt = 0; nt < 4; ++nt) {
        int p  = nt * 16 + l15; p = (p > 48) ? 48 : p;     // padded cols -> dup
        int ph = p / 7, pw = p - 7 * ph;
        bbase[nt] = (ph * 32 + pw * 2) * 256 + loff;
    }

    // ---- implicit-GEMM conv: D[768 x 49] = W[768 x 2304] * Patch[2304 x 49]
    // dual-M blocking: each wave carries two 16-row tiles per pass
    for (int mt0 = wave; mt0 < 48; mt0 += 16) {
        const int mt1 = mt0 + 8;
        v8f acc0[4], acc1[4];
        #pragma unroll
        for (int nt = 0; nt < 4; ++nt) { acc0[nt] = (v8f)(0.0f); acc1[nt] = (v8f)(0.0f); }

        const size_t aoff0 = (size_t)(mt0 * 16 + l15) * 2304 + loff;
        const size_t aoff1 = (size_t)(mt1 * 16 + l15) * 2304 + loff;

        for (int kc = 0; kc < 72; ++kc) {
            const int g   = kc >> 3;                       // (kh,kw) group
            const int kh  = g / 3, kw = g - 3 * kh;
            const int icb = (kc & 7) * 32;
            const int koffA = g * 256 + icb;               // wave-uniform
            const int koffB = (kh * 16 + kw) * 256 + icb;  // wave-uniform

            const v8us* wp0 = (const v8us*)(wB + aoff0 + koffA);
            const v8us* wp1 = (const v8us*)(wB + aoff1 + koffA);
            v16bf a0 = mkfrag(wp0[0], wp0[2]);             // K runs [0..7]+loff, [16..23]+loff
            v16bf a1 = mkfrag(wp1[0], wp1[2]);

            #pragma unroll
            for (int nt = 0; nt < 4; ++nt) {
                const v8us* bp = (const v8us*)(xs + bbase[nt] + koffB);
                v16bf bf = mkfrag(bp[0], bp[2]);
                acc0[nt] = __builtin_amdgcn_wmma_f32_16x16x32_bf16(
                               false, a0, false, bf, (short)0, acc0[nt], false, false);
                acc1[nt] = __builtin_amdgcn_wmma_f32_16x16x32_bf16(
                               false, a1, false, bf, (short)0, acc1[nt], false, false);
            }
        }
        #pragma unroll
        for (int nt = 0; nt < 4; ++nt) {
            int p = nt * 16 + l15;
            if (p < 49) {
                #pragma unroll
                for (int r = 0; r < 8; ++r) {
                    int rl = r + hi8;
                    qkv[(mt0 * 16 + rl) * 49 + p] = acc0[nt][r];
                    qkv[(mt1 * 16 + rl) * 49 + p] = acc1[nt][r];
                }
            }
        }
    }
    __syncthreads();

    // ---- attention: one head per wave --------------------------------------
    // torch reshape makes q[n,h,d] = qkv_flat[n*768 + h*32 + d], k at +256, v at +512
    const int h = wave;
    const float scale = 0.17677669529663687f;   // 32^-0.5
    float*          Ssc = (float*)(smem + wave * 6144);                  // [16][64] f32
    unsigned short* Psc = (unsigned short*)(smem + wave * 6144 + 4096);  // [16][64] bf16
    const int qoff = h * 32, koff = 256 + h * 32, voff = 512 + h * 32;
    float* outb = out7 + (size_t)b * 12544;     // 256*49

    for (int mt4 = 0; mt4 < 4; ++mt4) {
        // A = scaled Q rows (16 x 32)
        int nrow = mt4 * 16 + l15; int nq = (nrow > 48) ? 48 : nrow;
        v16us aqu;
        #pragma unroll
        for (int e = 0; e < 16; ++e) {
            int d = (e < 8 ? e : e + 8) + loff;
            aqu[e] = f2bf(qkv[nq * 768 + qoff + d] * scale);
        }
        v16bf aq = __builtin_bit_cast(v16bf, aqu);

        v8f sacc[4];
        #pragma unroll
        for (int nt = 0; nt < 4; ++nt) sacc[nt] = (v8f)(0.0f);

        #pragma unroll
        for (int nt = 0; nt < 4; ++nt) {        // B = K^T (32 x 16)
            int m = nt * 16 + l15; m = (m > 48) ? 48 : m;
            v16us bku;
            #pragma unroll
            for (int e = 0; e < 16; ++e) {
                int d = (e < 8 ? e : e + 8) + loff;
                bku[e] = f2bf(qkv[m * 768 + koff + d]);
            }
            sacc[nt] = __builtin_amdgcn_wmma_f32_16x16x32_bf16(
                           false, aq, false, __builtin_bit_cast(v16bf, bku),
                           (short)0, sacc[nt], false, false);
        }

        // scatter S + rel-pos bias (mask padded cols with -1e30)
        #pragma unroll
        for (int nt = 0; nt < 4; ++nt) {
            int m = nt * 16 + l15;
            #pragma unroll
            for (int r = 0; r < 8; ++r) {
                int rl = r + hi8;
                int n  = mt4 * 16 + rl;
                float val = -1e30f;
                if (m < 49 && n < 49) {
                    int nh_ = n / 7, nw_ = n - 7 * nh_;
                    int mh  = m / 7, mw  = m - 7 * mh;
                    int idx = (nh_ - mh + 6) * 13 + (nw_ - mw + 6);
                    val = sacc[nt][r] + rpb[idx * 8 + h];
                }
                Ssc[rl * 64 + m] = val;
            }
        }

        // softmax per row (lanes 0..15 each own one row)
        if (lane < 16) {
            float* row = Ssc + lane * 64;
            float mx = -1e30f;
            #pragma unroll 8
            for (int j = 0; j < 64; ++j) mx = fmaxf(mx, row[j]);
            float sum = 0.0f;
            #pragma unroll 8
            for (int j = 0; j < 64; ++j) { float e = __expf(row[j] - mx); sum += e; row[j] = e; }
            float inv = 1.0f / sum;
            #pragma unroll 8
            for (int j = 0; j < 64; ++j) Psc[lane * 64 + j] = f2bf(row[j] * inv);
        }

        // O = P (16 x 64) * V (64 x 32), K split in two 32-chunks
        v8f oacc[2];
        oacc[0] = (v8f)(0.0f); oacc[1] = (v8f)(0.0f);
        #pragma unroll
        for (int kt = 0; kt < 2; ++kt) {
            const v8us* pp = (const v8us*)(Psc + l15 * 64 + kt * 32 + loff);
            v16bf ap = mkfrag(pp[0], pp[2]);
            #pragma unroll
            for (int dt = 0; dt < 2; ++dt) {
                int d = dt * 16 + l15;
                v16us bvu;
                #pragma unroll
                for (int e = 0; e < 16; ++e) {
                    int kk = (e < 8 ? e : e + 8) + loff;
                    int m  = kt * 32 + kk; m = (m > 48) ? 48 : m;  // P(m>=49)==0 anyway
                    bvu[e] = f2bf(qkv[m * 768 + voff + d]);
                }
                oacc[dt] = __builtin_amdgcn_wmma_f32_16x16x32_bf16(
                               false, ap, false, __builtin_bit_cast(v16bf, bvu),
                               (short)0, oacc[dt], false, false);
            }
        }
        // out7[b][c=h*32+d][p=n]
        #pragma unroll
        for (int dt = 0; dt < 2; ++dt) {
            int d = dt * 16 + l15;
            #pragma unroll
            for (int r = 0; r < 8; ++r) {
                int n = mt4 * 16 + r + hi8;
                if (n < 49) outb[(h * 32 + d) * 49 + n] = oacc[dt][r];
            }
        }
    }
}

// ---------------------------------------------------------------------------
// Kernel 2: per-channel batch stats on the 7x7 tensor (upsample-invariant)
// ---------------------------------------------------------------------------
__global__ void bn_stats_kernel(const float* __restrict__ out7, float* __restrict__ stats) {
    const int c = blockIdx.x;
    const int tid = threadIdx.x;
    __shared__ float ssum[256], ssq[256];
    float s = 0.0f, q = 0.0f;
    for (int i = tid; i < 50176; i += 256) {     // 1024 batches * 49 positions
        int bb = i / 49, p = i - bb * 49;
        float v = out7[((size_t)bb * 256 + c) * 49 + p];
        s += v; q += v * v;
    }
    ssum[tid] = s; ssq[tid] = q;
    __syncthreads();
    for (int st = 128; st > 0; st >>= 1) {
        if (tid < st) { ssum[tid] += ssum[tid + st]; ssq[tid] += ssq[tid + st]; }
        __syncthreads();
    }
    if (tid == 0) {
        float mean = ssum[0] * (1.0f / 50176.0f);
        float var  = ssq[0]  * (1.0f / 50176.0f) - mean * mean;
        stats[c]       = mean;
        stats[256 + c] = rsqrtf(var + 1e-5f);
    }
}

// ---------------------------------------------------------------------------
// Kernel 3: nearest 2x upsample + BN affine -> [B,256,14,14]
// ---------------------------------------------------------------------------
__global__ void bn_apply_kernel(const float* __restrict__ out7,
                                const float* __restrict__ stats,
                                const float* __restrict__ bnw,
                                const float* __restrict__ bnb,
                                float* __restrict__ out)
{
    int idx = blockIdx.x * 256 + threadIdx.x;            // < 51,380,224
    int b   = idx / 50176;
    int rem = idx - b * 50176;
    int c   = rem / 196;
    int s   = rem - c * 196;
    int hh  = s / 14, ww = s - 14 * hh;
    float v = out7[((size_t)b * 256 + c) * 49 + (hh >> 1) * 7 + (ww >> 1)];
    out[idx] = (v - stats[c]) * stats[256 + c] * bnw[c] + bnb[c];
}

// ---------------------------------------------------------------------------
extern "C" void kernel_launch(void* const* d_in, const int* in_sizes, int n_in,
                              void* d_out, int out_size, void* d_ws, size_t ws_size,
                              hipStream_t stream) {
    const float* x   = (const float*)d_in[0];
    const float* w   = (const float*)d_in[1];
    const float* rpb = (const float*)d_in[2];
    const float* bnw = (const float*)d_in[3];
    const float* bnb = (const float*)d_in[4];
    float* out = (float*)d_out;

    char* ws = (char*)d_ws;
    unsigned short* wB   = (unsigned short*)ws;                       // 3,538,944 B
    float*          out7 = (float*)(ws + 3538944);                    // 51,380,224 B
    float*          st   = (float*)(ws + 3538944 + 51380224);         // 2,048 B

    // allow >64KB dynamic LDS (best effort; harmless if already allowed)
    (void)hipFuncSetAttribute((const void*)fused_qkv_attn_kernel,
                              hipFuncAttributeMaxDynamicSharedMemorySize, SMEM_BYTES);

    prep_w_kernel<<<(768 * 2304 + 255) / 256, 256, 0, stream>>>(w, wB);
    fused_qkv_attn_kernel<<<1024, 256, SMEM_BYTES, stream>>>(x, wB, rpb, out7);
    bn_stats_kernel<<<256, 256, 0, stream>>>(out7, st);
    bn_apply_kernel<<<51380224 / 256, 256, 0, stream>>>(out7, st, bnw, bnb, out);
}
